// NodeCycleFeatures_36481452212243
// MI455X (gfx1250) — matrix operand order repaired and checked by reading, not compiled, not run on hardware
//
#include <hip/hip_runtime.h>
#include <hip/hip_bf16.h>

typedef __attribute__((ext_vector_type(16))) _Float16 v16h;
typedef __attribute__((ext_vector_type(8)))  _Float16 v8h;
typedef __attribute__((ext_vector_type(2)))  _Float16 h2;
typedef __attribute__((ext_vector_type(8)))  float    v8f;

#define NN   64
#define AP   72   // f16 pitch: 144 bytes/row (16B aligned, conflict-free spread)
#define KP   68   // f32 pitch: 272 bytes/row (16B aligned)

__device__ __forceinline__ v16h load_afrag(const _Float16* __restrict__ src,
                                           int row, int kb, int h)
{
    v16h a;
    ((v8h*)&a)[0] = *(const v8h*)(src + row*AP + kb + h*16);
    ((v8h*)&a)[1] = *(const v8h*)(src + row*AP + kb + h*16 + 8);
    return a;
}

__global__ __launch_bounds__(256)
void node_cycle_kernel(const float* __restrict__ E,
                       const float* __restrict__ node_mask,
                       float* __restrict__ out, int B)
{
    __shared__ __align__(16) _Float16 Ah [NN*AP];
    __shared__ __align__(16) _Float16 K2h[NN*AP];
    __shared__ __align__(16) float    K3f[NN*KP];
    __shared__ float dL[NN], c3L[NN], da4L[NN], da5L[NN], d2L[NN];
    __shared__ float AdL[NN], AtL[NN], t1L[NN], t3L[NN], t8L[NN];
    __shared__ float c4L[NN], c5L[NN];

    const int b    = blockIdx.x;
    const int t    = threadIdx.x;
    const int lane = t & 31;
    const int wave = t >> 5;

    // ---------- Phase 0: A = E[b,:,:,1] -> LDS f16 (values exactly 0/1) ----------
    {
        const float4* E4 = (const float4*)(E + (size_t)b * (NN*NN*2));
#pragma unroll
        for (int r = 0; r < 8; ++r) {
            int idx4 = r * 256 + t;        // float4 = 2 elements x 2 channels
            float4 v = E4[idx4];
            int e  = idx4 * 2;             // even element index in [0,4096)
            int i0 = e >> 6, j0 = e & 63;  // j0 even -> pair stays in-row
            h2 p; p[0] = (_Float16)v.y; p[1] = (_Float16)v.w;
            *(h2*)(Ah + i0*AP + j0) = p;   // one ds_store_b32
        }
    }
    __syncthreads();

    // Tile pairing: wave handles (ti0,tj) and (ti0+2,tj) -> shared B fragment.
    const int n   = lane & 15;
    const int h   = lane >> 4;
    const int tj  = wave & 3;
    const int ti0 = wave >> 2;        // 0 or 1
    const int ti1 = ti0 + 2;          // 2 or 3
    const int brow = tj*16 + n;       // B frag row (via symmetry) & transposed-store row

    // ---------- Phase 1: k2 = A @ A  (entries <= 63, f16-exact) ----------
    {
        v8f acc0 = {0,0,0,0,0,0,0,0}, acc1 = {0,0,0,0,0,0,0,0};
#pragma unroll
        for (int kk = 0; kk < 2; ++kk) {
            const int kb = kk * 32;
            v16h bf;
            ((v8h*)&bf)[0] = *(const v8h*)(Ah + brow*AP + kb + h*16);
            ((v8h*)&bf)[1] = *(const v8h*)(Ah + brow*AP + kb + h*16 + 8);
            v16h a0 = load_afrag(Ah, ti0*16 + n, kb, h);
            v16h a1 = load_afrag(Ah, ti1*16 + n, kb, h);
            acc0 = __builtin_amdgcn_wmma_f32_16x16x32_f16(false, a0, false, bf,
                                                          (short)0, acc0, false, false);
            acc1 = __builtin_amdgcn_wmma_f32_16x16x32_f16(false, a1, false, bf,
                                                          (short)0, acc1, false, false);
        }
        // k2 is exactly symmetric -> store tiles transposed: lane's 8 values are
        // contiguous f16 -> one ds_store_b128 per tile.
        v8h p0, p1;
#pragma unroll
        for (int v = 0; v < 8; ++v) { p0[v] = (_Float16)acc0[v]; p1[v] = (_Float16)acc1[v]; }
        *(v8h*)(K2h + brow*AP + ti0*16 + 8*h) = p0;
        *(v8h*)(K2h + brow*AP + ti1*16 + 8*h) = p1;
    }
    __syncthreads();

    // ---------- Phase 2: k3 = k2 @ A  (f32, exact integers) ----------
    {
        v8f acc0 = {0,0,0,0,0,0,0,0}, acc1 = {0,0,0,0,0,0,0,0};
#pragma unroll
        for (int kk = 0; kk < 2; ++kk) {
            const int kb = kk * 32;
            v16h bf;
            ((v8h*)&bf)[0] = *(const v8h*)(Ah + brow*AP + kb + h*16);
            ((v8h*)&bf)[1] = *(const v8h*)(Ah + brow*AP + kb + h*16 + 8);
            v16h a0 = load_afrag(K2h, ti0*16 + n, kb, h);
            v16h a1 = load_afrag(K2h, ti1*16 + n, kb, h);
            acc0 = __builtin_amdgcn_wmma_f32_16x16x32_f16(false, a0, false, bf,
                                                          (short)0, acc0, false, false);
            acc1 = __builtin_amdgcn_wmma_f32_16x16x32_f16(false, a1, false, bf,
                                                          (short)0, acc1, false, false);
        }
        // k3 symmetric -> transposed store: 8 contiguous f32 = 2x ds_store_b128.
        float* r0 = K3f + brow*KP + ti0*16 + 8*h;
        float* r1 = K3f + brow*KP + ti1*16 + 8*h;
        *(float4*)(r0)     = (float4){acc0[0], acc0[1], acc0[2], acc0[3]};
        *(float4*)(r0 + 4) = (float4){acc0[4], acc0[5], acc0[6], acc0[7]};
        *(float4*)(r1)     = (float4){acc1[0], acc1[1], acc1[2], acc1[3]};
        *(float4*)(r1 + 4) = (float4){acc1[4], acc1[5], acc1[6], acc1[7]};
    }
    __syncthreads();

    // ---------- Phase 3: per-row reductions (4 threads per row) ----------
    {
        const int i = t >> 2, q = t & 3, jb = q * 16;
        v8h a0 = *(const v8h*)(Ah  + i*AP + jb);
        v8h a1 = *(const v8h*)(Ah  + i*AP + jb + 8);
        v8h b0 = *(const v8h*)(K2h + i*AP + jb);
        v8h b1 = *(const v8h*)(K2h + i*AP + jb + 8);
        float a16[16], k2v[16], k3v[16];
#pragma unroll
        for (int e = 0; e < 8; ++e) {
            a16[e] = (float)a0[e]; a16[8+e] = (float)a1[e];
            k2v[e] = (float)b0[e]; k2v[8+e] = (float)b1[e];
        }
        const float4* K34 = (const float4*)(K3f + i*KP + jb);
#pragma unroll
        for (int c = 0; c < 4; ++c) {
            float4 v = K34[c];
            k3v[c*4+0]=v.x; k3v[c*4+1]=v.y; k3v[c*4+2]=v.z; k3v[c*4+3]=v.w;
        }
        float dP=0, c3P=0, da4P=0, da5P=0, t3P=0, t1P=0, t8P=0;
#pragma unroll
        for (int e = 0; e < 16; ++e) {
            float a = a16[e], x2 = k2v[e], x3 = k3v[e];
            dP  += a;        c3P  += a*x2;    da4P += x2*x2;
            da5P+= x2*x3;    t3P  += a*x2*x2; t1P  += x3*x3;  t8P += x3;
        }
#pragma unroll
        for (int off = 1; off < 4; off <<= 1) {
            dP   += __shfl_xor(dP,   off); c3P  += __shfl_xor(c3P,  off);
            da4P += __shfl_xor(da4P, off); da5P += __shfl_xor(da5P, off);
            t3P  += __shfl_xor(t3P,  off); t1P  += __shfl_xor(t1P,  off);
            t8P  += __shfl_xor(t8P,  off);
        }
        if (q == 0) {
            dL[i]=dP; c3L[i]=c3P; da4L[i]=da4P; da5L[i]=da5P;
            t3L[i]=t3P; t1L[i]=t1P; t8L[i]=t8P;
            d2L[i] = (float)K2h[i*AP + i];
        }
    }
    __syncthreads();

    // ---------- Phase 4: (A d)_i and (A c3)_i ----------
    {
        const int i = t >> 2, q = t & 3, jb = q * 16;
        v8h a0 = *(const v8h*)(Ah + i*AP + jb);
        v8h a1 = *(const v8h*)(Ah + i*AP + jb + 8);
        float AdP = 0.f, AtP = 0.f;
#pragma unroll
        for (int e = 0; e < 8; ++e) {
            float aa = (float)a0[e];
            AdP += aa * dL[jb+e];   AtP += aa * c3L[jb+e];
            float ab = (float)a1[e];
            AdP += ab * dL[jb+8+e]; AtP += ab * c3L[jb+8+e];
        }
#pragma unroll
        for (int off = 1; off < 4; off <<= 1) {
            AdP += __shfl_xor(AdP, off); AtP += __shfl_xor(AtP, off);
        }
        if (q == 0) { AdL[i] = AdP; AtL[i] = AtP; }
    }
    __syncthreads();

    // ---------- Phase 5a: per-node cycle counts + x outputs ----------
    if (t < NN) {
        const int i = t;
        float d  = dL[i], c3 = c3L[i];
        float c4 = da4L[i] - d*(d - 1.0f) - AdL[i];
        float c5 = da5L[i] - 2.0f*c3*d - AtL[i] + c3;
        c4L[i] = c4; c5L[i] = c5;
        float m = node_mask[(size_t)b*NN + i];
        float* xo = out + (size_t)b*(NN*3) + i*3;
        xo[0] = fminf(((c3*0.5f)*m)/10.0f, 1.0f);
        xo[1] = fminf(((c4*0.5f)*m)/10.0f, 1.0f);
        xo[2] = fminf(((c5*0.5f)*m)/10.0f, 1.0f);
    }
    __syncthreads();

    // ---------- Phase 5b: scalar reductions (deterministic shuffle tree) ----------
    if (wave == 0) {
        const int l = lane;
        float c3a = c3L[l],  c3b = c3L[l+32];
        float d2a = d2L[l],  d2b = d2L[l+32];
        float w4a = da4L[l], w4b = da4L[l+32];
        float s[12];
        s[0]  = c3a + c3b;                      // sum c3 (t6)
        s[1]  = c4L[l] + c4L[l+32];             // sum c4
        s[2]  = c5L[l] + c5L[l+32];             // sum c5
        s[3]  = t1L[l] + t1L[l+32];             // t1 = trace(k6)
        s[4]  = c3a*c3a + c3b*c3b;              // t2
        s[5]  = t3L[l] + t3L[l+32];             // t3
        s[6]  = d2a*w4a + d2b*w4b;              // t4
        s[7]  = w4a + w4b;                      // t5
        s[8]  = d2a*d2a*d2a + d2b*d2b*d2b;      // t7
        s[9]  = t8L[l] + t8L[l+32];             // t8
        s[10] = d2a*d2a + d2b*d2b;              // t9
        s[11] = d2a + d2b;                      // t10
#pragma unroll
        for (int off = 16; off >= 1; off >>= 1) {
#pragma unroll
            for (int k = 0; k < 12; ++k) s[k] += __shfl_xor(s[k], off);
        }
        if (lane == 0) {
            float c6 = s[3] - 3.f*s[4] + 9.f*s[5] - 6.f*s[6] + 6.f*s[7]
                     - 4.f*s[0] + 4.f*s[8] + 3.f*s[9] - 12.f*s[10] + 4.f*s[11];
            float* yo = out + (size_t)B*(NN*3) + (size_t)b*4;
            yo[0] = fminf((s[0]/6.0f)/10.0f,  1.0f);
            yo[1] = fminf((s[1]/8.0f)/10.0f,  1.0f);
            yo[2] = fminf((s[2]/10.0f)/10.0f, 1.0f);
            yo[3] = fminf((c6/12.0f)/10.0f,   1.0f);
        }
    }
}

extern "C" void kernel_launch(void* const* d_in, const int* in_sizes, int n_in,
                              void* d_out, int out_size, void* d_ws, size_t ws_size,
                              hipStream_t stream)
{
    const float* E    = (const float*)d_in[0];
    const float* mask = (const float*)d_in[1];
    float* out        = (float*)d_out;
    const int B = in_sizes[1] / NN;   // node_mask is [B, 64]
    node_cycle_kernel<<<B, 256, 0, stream>>>(E, mask, out, B);
}
